// Trans_Semantics_65635690218073
// MI455X (gfx1250) — matrix operand order repaired and checked by reading, not compiled, not run on hardware
//
#include <hip/hip_runtime.h>
#include <math.h>

// ---------------- model dims ----------------
#define F_   128
#define W_   9
#define B_   16
#define E_   6
#define NH_  3
#define NL_  3
#define FFD_ 12
#define HID_ 768          // F*E
#define S_   1152         // W*F
#define TOPM_ 99
#define G3_  2304         // 3*HID
#define KT_  (HID_ / 32)  // 24 k-tiles for the GRU GEMMs

typedef __attribute__((ext_vector_type(2)))  float  v2f;
typedef __attribute__((ext_vector_type(8)))  float  v8f;
typedef __attribute__((ext_vector_type(16))) __bf16 v16bf;

// ---------------- workspace layout (bytes) ----------------
#define OFF_X    0ull          // f32 [B][S][8]       589824
#define OFF_Q    589824ull     // f32 [B][NH][S][4]   884736
#define OFF_K    1474560ull
#define OFF_V    2359296ull
#define OFF_O    3244032ull    // f32 [B][S][8]       589824
#define OFF_GI   3833856ull    // f32 [16][2304]      147456
#define OFF_GH   3981312ull
#define OFF_H    4128768ull    // f32 [16][768]        49152
#define OFF_YBF  4177920ull    // bf16 [8][24][32][16] swizzled  196608
#define OFF_WIHT 4374528ull    // bf16 [144][24][32][16] swizzled 3538944
#define OFF_WHHT 7913472ull
#define OFF_FCNW 11452416ull   // bf16 [8][24][32][16] swizzled   196608
#define OFF_HBF  11649024ull   // bf16 [24][32][16] swizzled       24576

// ---------------- helpers ----------------
__device__ __forceinline__ unsigned short f2bf(float f) {
  unsigned u = __float_as_uint(f);
  if ((u & 0x7fffffffu) > 0x7f800000u) return (unsigned short)0x7fc0u;
  unsigned r = u + 0x7fffu + ((u >> 16) & 1u);
  return (unsigned short)(r >> 16);
}
__device__ __forceinline__ unsigned ordf(float f) {
  unsigned u = __float_as_uint(f);
  return (u & 0x80000000u) ? ~u : (u | 0x80000000u);
}
__device__ __forceinline__ float sigmf(float x) { return 1.f / (1.f + __expf(-x)); }

// WMMA bf16 fragment swizzles (ISA 7.12.2 VGPR layouts).
// A (16x32, lane m = M row):   frag element for (m, k): lane = m + 16*((r>>3)&1),
//                              idx = ((r&16)>>1)|(r&7), r = k%32.
__device__ __forceinline__ size_t a_sw(int m, int k) {
  int kt = k >> 5, r = k & 31;
  int lane = m + ((r & 8) << 1);
  int idx  = ((r & 16) >> 1) | (r & 7);
  return (((size_t)kt << 5) + lane) * 16 + idx;
}
// B (32x16, lane n = N col):   frag element for (k, n): lane = (n&15) + 16*((r>>3)&1),
//                              idx = ((r&16)>>1)|(r&7); tiles indexed [nt][kt].
__device__ __forceinline__ size_t b_sw(int k, int n, int nkt) {
  int nt = n >> 4, kt = k >> 5, r = k & 31;
  int lane = (n & 15) + ((r & 8) << 1);
  int idx  = ((r & 16) >> 1) | (r & 7);
  return ((((size_t)nt * nkt + kt) << 5) + lane) * 16 + idx;
}

__device__ __forceinline__ void ln6(const float* in, float* out,
                                    const float* s, const float* b) {
  float m = 0.f;
  #pragma unroll
  for (int j = 0; j < E_; ++j) m += in[j];
  m *= (1.f / (float)E_);
  float v = 0.f;
  #pragma unroll
  for (int j = 0; j < E_; ++j) { float d = in[j] - m; v += d * d; }
  v *= (1.f / (float)E_);
  float inv = rsqrtf(v + 1e-5f);
  #pragma unroll
  for (int j = 0; j < E_; ++j) out[j] = (in[j] - m) * inv * s[j] + b[j];
}

// ---------------- weight conversion: f32 -> bf16, pre-swizzled to WMMA frags ----------------
__global__ void k_conv_gru(const float* __restrict__ Wih, const float* __restrict__ Whh,
                           unsigned short* __restrict__ WihSW, unsigned short* __restrict__ WhhSW) {
  int idx = blockIdx.x * 256 + threadIdx.x;
  if (idx >= G3_ * HID_) return;
  int n = idx / HID_;        // gate/output column (0..2303)
  int k = idx % HID_;        // contraction index
  size_t d = b_sw(k, n, KT_);
  WihSW[d] = f2bf(Wih[idx]);
  WhhSW[d] = f2bf(Whh[idx]);
}
__global__ void k_conv_fcn(const float* __restrict__ fw, unsigned short* __restrict__ fwsw) {
  int idx = blockIdx.x * 256 + threadIdx.x;
  if (idx >= HID_ * F_) return;
  int k = idx / F_, n = idx % F_;     // fcn_W is [HID][F] => B[k][n]
  fwsw[b_sw(k, n, KT_)] = f2bf(fw[idx]);
}

// ---------------- AutoDis + pos emb -> x[B][S][8] ----------------
__global__ void k_autodis(const float* __restrict__ src,
                          const float* __restrict__ w1w, const float* __restrict__ w1b,
                          const float* __restrict__ w2w, const float* __restrict__ w2b,
                          const float* __restrict__ meta, const float* __restrict__ pos,
                          float* __restrict__ x) {
  int tid = blockIdx.x * 256 + threadIdx.x;
  if (tid >= W_ * B_ * F_) return;
  int w = tid / (B_ * F_);
  int r = tid % (B_ * F_);
  int b = r / F_, f = r % F_;
  float v = src[(w * B_ + b) * F_ + f];
  float h1[E_], h2[E_], t6[E_];
  #pragma unroll
  for (int e = 0; e < E_; ++e) {
    float t = v * w1w[f * E_ + e] + w1b[f * E_ + e];
    h1[e] = (t >= 0.f) ? t : 0.01f * t;
  }
  #pragma unroll
  for (int j = 0; j < E_; ++j) {
    float s = w2b[f * E_ + j];
    #pragma unroll
    for (int kk = 0; kk < E_; ++kk) s += h1[kk] * w2w[(f * E_ + kk) * E_ + j];
    h2[j] = s;
  }
  float mx = -3.0e38f;
  #pragma unroll
  for (int j = 0; j < E_; ++j) { t6[j] = (h2[j] + 0.5f * h1[j]) * 1e5f; mx = fmaxf(mx, t6[j]); }
  float se = 0.f;
  #pragma unroll
  for (int j = 0; j < E_; ++j) { t6[j] = __expf(t6[j] - mx); se += t6[j]; }
  float inv = 1.f / se;
  int sidx = w * F_ + f;
  float* xp = x + ((size_t)b * S_ + sidx) * 8;
  #pragma unroll
  for (int e = 0; e < E_; ++e) {
    float s = 0.f;
    #pragma unroll
    for (int kk = 0; kk < E_; ++kk) s += t6[kk] * meta[(f * E_ + kk) * E_ + e];
    xp[e] = s * inv * 2.449489742783178f + pos[sidx * E_ + e];   // *sqrt(6)
  }
  xp[6] = 0.f; xp[7] = 0.f;
}

// ---------------- QKV projection (per layer) ----------------
__global__ void k_qkv(const float* __restrict__ x,
                      const float* __restrict__ Wq, const float* __restrict__ Wk,
                      const float* __restrict__ Wv,
                      const float* __restrict__ bq, const float* __restrict__ bk,
                      const float* __restrict__ bv,
                      float* __restrict__ q, float* __restrict__ k, float* __restrict__ v) {
  int tid = blockIdx.x * 256 + threadIdx.x;
  if (tid >= B_ * S_) return;
  int b = tid / S_, s = tid % S_;
  float xv[E_];
  #pragma unroll
  for (int e = 0; e < E_; ++e) xv[e] = x[((size_t)b * S_ + s) * 8 + e];
  const float scl = 0.7071067811865476f;   // 1/sqrt(DH) folded into q
  #pragma unroll
  for (int j = 0; j < E_; ++j) {
    float qj = bq[j], kj = bk[j], vj = bv[j];
    #pragma unroll
    for (int e = 0; e < E_; ++e) {
      qj += xv[e] * Wq[e * E_ + j];
      kj += xv[e] * Wk[e * E_ + j];
      vj += xv[e] * Wv[e * E_ + j];
    }
    int h = j >> 1, d = j & 1;
    size_t base = ((size_t)(b * NH_ + h) * S_ + s) * 4;
    q[base + d] = qj * scl;  k[base + d] = kj;  v[base + d] = vj;
    if (d == 1) {            // zero the K-pad (d=2,3)
      q[base + 2] = 0.f; q[base + 3] = 0.f;
      k[base + 2] = 0.f; k[base + 3] = 0.f;
      v[base + 2] = 0.f; v[base + 3] = 0.f;
    }
  }
}

// ---------------- attention: WMMA f32 16x16x4 scores + top-m + softmax + PV ----------------
__global__ void k_attn(const float* __restrict__ q, const float* __restrict__ k,
                       const float* __restrict__ v, float* __restrict__ o) {
  extern __shared__ float sc[];                 // [16][S_]
  int blk = blockIdx.x;
  int qt = blk % (S_ / 16);
  int bh = blk / (S_ / 16);
  int b = bh / NH_, h = bh % NH_;
  int q0 = qt * 16;
  int lane = threadIdx.x & 31;
  int wv   = threadIdx.x >> 5;                  // wave 0..7
  int m = lane & 15, half = lane >> 4;

  const float* qb = q + ((size_t)(b * NH_ + h) * S_) * 4;
  const float* kb = k + ((size_t)(b * NH_ + h) * S_) * 4;
  const float* vb = v + ((size_t)(b * NH_ + h) * S_) * 4;

  // A fragment: 16x4 f32, lane m = row, half selects K pair
  v2f a;
  a.x = qb[(q0 + m) * 4 + 2 * half];
  a.y = qb[(q0 + m) * 4 + 2 * half + 1];

  // uniform trip count (9 iters); kt wave-uniform => EXEC stays all-ones
  for (int it = 0; it < (S_ / 16) / 8; ++it) {
    int kt = it * 8 + wv;
    v2f bf;
    int col = kt * 16 + m;                      // B: lane m = N column
    bf.x = kb[col * 4 + 2 * half];
    bf.y = kb[col * 4 + 2 * half + 1];
    v8f c = {0.f, 0.f, 0.f, 0.f, 0.f, 0.f, 0.f, 0.f};
    c = __builtin_amdgcn_wmma_f32_16x16x4_f32(false, a, false, bf,
                                              (short)0, c, false, false);
    #pragma unroll
    for (int i = 0; i < 8; ++i)
      sc[(i + 8 * half) * S_ + kt * 16 + m] = c[i];
  }
  __syncthreads();

  for (int rr = 0; rr < 2; ++rr) {
    int row = wv * 2 + rr;
    float* rp = sc + row * S_;

    // radix-select the TOPM-th largest (greedy bit-build on ordered uint)
    unsigned prefix = 0u;
    for (int bit = 31; bit >= 0; --bit) {
      unsigned test = prefix | (1u << bit);
      int cnt = 0;
      for (int c2 = lane; c2 < S_; c2 += 32) cnt += (ordf(rp[c2]) >= test) ? 1 : 0;
      #pragma unroll
      for (int off = 16; off > 0; off >>= 1) cnt += __shfl_xor(cnt, off, 32);
      if (cnt >= TOPM_) prefix = test;
    }

    float mx = -3.0e38f;
    for (int c2 = lane; c2 < S_; c2 += 32) mx = fmaxf(mx, rp[c2]);
    #pragma unroll
    for (int off = 16; off > 0; off >>= 1) mx = fmaxf(mx, __shfl_xor(mx, off, 32));

    float sum = 0.f;
    for (int c2 = lane; c2 < S_; c2 += 32) {
      float val = rp[c2];
      float e2 = (ordf(val) >= prefix) ? __expf(val - mx) : 0.f;
      rp[c2] = e2;
      sum += e2;
    }
    #pragma unroll
    for (int off = 16; off > 0; off >>= 1) sum += __shfl_xor(sum, off, 32);
    float inv = 1.f / sum;

    float o0 = 0.f, o1 = 0.f;
    for (int c2 = lane; c2 < S_; c2 += 32) {
      float p = rp[c2] * inv;
      o0 += p * vb[c2 * 4 + 0];
      o1 += p * vb[c2 * 4 + 1];
    }
    #pragma unroll
    for (int off = 16; off > 0; off >>= 1) {
      o0 += __shfl_xor(o0, off, 32);
      o1 += __shfl_xor(o1, off, 32);
    }
    if (lane == 0) {
      float* op = o + ((size_t)b * S_ + q0 + row) * 8 + h * 2;
      op[0] = o0; op[1] = o1;
    }
  }
}

// ---------------- Wo + LN1 + FFN + LN2 (per layer, in-place on x) ----------------
__global__ void k_post(float* __restrict__ x, const float* __restrict__ o,
                       const float* __restrict__ Wo, const float* __restrict__ bo,
                       const float* __restrict__ l1s, const float* __restrict__ l1b,
                       const float* __restrict__ W1, const float* __restrict__ b1,
                       const float* __restrict__ W2, const float* __restrict__ b2,
                       const float* __restrict__ l2s, const float* __restrict__ l2b) {
  int tid = blockIdx.x * 256 + threadIdx.x;
  if (tid >= B_ * S_) return;
  int b = tid / S_, s = tid % S_;
  float xv[E_], ov[E_], t[E_];
  #pragma unroll
  for (int e = 0; e < E_; ++e) {
    xv[e] = x[((size_t)b * S_ + s) * 8 + e];
    ov[e] = o[((size_t)b * S_ + s) * 8 + e];
  }
  #pragma unroll
  for (int j = 0; j < E_; ++j) {
    float a = bo[j];
    #pragma unroll
    for (int e = 0; e < E_; ++e) a += ov[e] * Wo[e * E_ + j];
    t[j] = xv[j] + a;
  }
  ln6(t, xv, l1s, l1b);
  float u[FFD_];
  #pragma unroll
  for (int jf = 0; jf < FFD_; ++jf) {
    float a = b1[jf];
    #pragma unroll
    for (int e = 0; e < E_; ++e) a += xv[e] * W1[e * FFD_ + jf];
    u[jf] = fmaxf(a, 0.f);
  }
  #pragma unroll
  for (int j = 0; j < E_; ++j) {
    float a = b2[j];
    #pragma unroll
    for (int jf = 0; jf < FFD_; ++jf) a += u[jf] * W2[jf * E_ + j];
    t[j] = xv[j] + a;
  }
  ln6(t, xv, l2s, l2b);
  #pragma unroll
  for (int e = 0; e < E_; ++e) x[((size_t)b * S_ + s) * 8 + e] = xv[e];
}

// ---------------- pack y (bf16, WMMA-A swizzled) for the GRU ----------------
__global__ void k_pack_y(const float* __restrict__ x, unsigned short* __restrict__ ysw) {
  int tid = blockIdx.x * 256 + threadIdx.x;
  if (tid >= (W_ - 1) * B_ * HID_) return;
  int w = tid / (B_ * HID_);
  int r = tid % (B_ * HID_);
  int b = r / HID_, j = r % HID_;
  int f = j / E_, e = j % E_;
  float val = x[((size_t)b * S_ + w * F_ + f) * 8 + e];
  ysw[(size_t)w * (B_ * HID_) + a_sw(b, j)] = f2bf(val);
}

__global__ void k_init_h(const float* __restrict__ ih, float* __restrict__ h,
                         unsigned short* __restrict__ hsw) {
  int tid = blockIdx.x * 256 + threadIdx.x;
  if (tid >= B_ * HID_) return;
  int b = tid / HID_, j = tid % HID_;
  float v = ih[tid];
  h[tid] = v;
  hsw[a_sw(b, j)] = f2bf(v);
}

// ---------------- GRU GEMM step: gi/gh[16][2304] via bf16 WMMA (b128 frag loads) ----------------
__global__ void __launch_bounds__(32)
k_gru_mm(const unsigned short* __restrict__ ysw, const unsigned short* __restrict__ hsw,
         const unsigned short* __restrict__ WihSW, const unsigned short* __restrict__ WhhSW,
         const float* __restrict__ bih, const float* __restrict__ bhh,
         float* __restrict__ gi, float* __restrict__ gh) {
  int nt = blockIdx.x;
  int lane = threadIdx.x & 31;
  int m = lane & 15, half = lane >> 4;
  const unsigned short* bi_base = WihSW + (size_t)nt * KT_ * 512;
  const unsigned short* bh_base = WhhSW + (size_t)nt * KT_ * 512;
  v8f ci0 = {0.f,0.f,0.f,0.f,0.f,0.f,0.f,0.f}, ci1 = ci0, ch0 = ci0, ch1 = ci0;
  #pragma unroll 4
  for (int kt = 0; kt < KT_; ++kt) {
    size_t foff = ((size_t)kt * 32 + lane) * 16;      // 32B-aligned fragment
    v16bf ay = *(const v16bf*)(ysw + foff);
    v16bf ah = *(const v16bf*)(hsw + foff);
    v16bf bi = *(const v16bf*)(bi_base + foff);
    v16bf bh = *(const v16bf*)(bh_base + foff);
    if (kt & 1) {
      ci1 = __builtin_amdgcn_wmma_f32_16x16x32_bf16(false, ay, false, bi, (short)0, ci1, false, false);
      ch1 = __builtin_amdgcn_wmma_f32_16x16x32_bf16(false, ah, false, bh, (short)0, ch1, false, false);
    } else {
      ci0 = __builtin_amdgcn_wmma_f32_16x16x32_bf16(false, ay, false, bi, (short)0, ci0, false, false);
      ch0 = __builtin_amdgcn_wmma_f32_16x16x32_bf16(false, ah, false, bh, (short)0, ch0, false, false);
    }
  }
  #pragma unroll
  for (int i = 0; i < 8; ++i) {
    int mm = i + 8 * half;       // batch row
    int nn = nt * 16 + m;        // gate column
    gi[mm * G3_ + nn] = ci0[i] + ci1[i] + bih[nn];
    gh[mm * G3_ + nn] = ch0[i] + ch1[i] + bhh[nn];
  }
}

// ---------------- GRU gate combine (torch r,z,n ordering) ----------------
__global__ void k_gru_comb(const float* __restrict__ gi, const float* __restrict__ gh,
                           float* __restrict__ h, unsigned short* __restrict__ hsw) {
  int tid = blockIdx.x * 256 + threadIdx.x;
  if (tid >= B_ * HID_) return;
  int b = tid / HID_, j = tid % HID_;
  const float* gib = gi + (size_t)b * G3_;
  const float* ghb = gh + (size_t)b * G3_;
  float r = sigmf(gib[j] + ghb[j]);
  float z = sigmf(gib[HID_ + j] + ghb[HID_ + j]);
  float n = tanhf(gib[2 * HID_ + j] + r * ghb[2 * HID_ + j]);
  float hv = h[tid];
  float hn = (1.f - z) * n + z * hv;
  h[tid] = hn;
  hsw[a_sw(b, j)] = f2bf(hn);
}

// ---------------- final head: sigmoid(h @ fcn_W + b) via bf16 WMMA ----------------
__global__ void __launch_bounds__(32)
k_final(const unsigned short* __restrict__ hsw, const unsigned short* __restrict__ fwsw,
        const float* __restrict__ fb, float* __restrict__ out) {
  int nt = blockIdx.x;
  int lane = threadIdx.x & 31;
  int m = lane & 15, half = lane >> 4;
  const unsigned short* b_base = fwsw + (size_t)nt * KT_ * 512;
  v8f c0 = {0.f,0.f,0.f,0.f,0.f,0.f,0.f,0.f}, c1 = c0;
  #pragma unroll 4
  for (int kt = 0; kt < KT_; ++kt) {
    size_t foff = ((size_t)kt * 32 + lane) * 16;
    v16bf a = *(const v16bf*)(hsw + foff);
    v16bf b = *(const v16bf*)(b_base + foff);
    if (kt & 1)
      c1 = __builtin_amdgcn_wmma_f32_16x16x32_bf16(false, a, false, b, (short)0, c1, false, false);
    else
      c0 = __builtin_amdgcn_wmma_f32_16x16x32_bf16(false, a, false, b, (short)0, c0, false, false);
  }
  #pragma unroll
  for (int i = 0; i < 8; ++i)
    out[(i + 8 * half) * F_ + nt * 16 + m] = sigmf(c0[i] + c1[i] + fb[nt * 16 + m]);
}

// ---------------- host launcher ----------------
extern "C" void kernel_launch(void* const* d_in, const int* in_sizes, int n_in,
                              void* d_out, int out_size, void* d_ws, size_t ws_size,
                              hipStream_t stream) {
  const float* src   = (const float*)d_in[0];
  const float* w1w   = (const float*)d_in[1];
  const float* w1b   = (const float*)d_in[2];
  const float* w2w   = (const float*)d_in[3];
  const float* w2b   = (const float*)d_in[4];
  const float* meta  = (const float*)d_in[5];
  const float* pos   = (const float*)d_in[6];
  const float* Wq    = (const float*)d_in[7];
  const float* Wk    = (const float*)d_in[8];
  const float* Wv    = (const float*)d_in[9];
  const float* Wo    = (const float*)d_in[10];
  const float* bq    = (const float*)d_in[11];
  const float* bk    = (const float*)d_in[12];
  const float* bv    = (const float*)d_in[13];
  const float* bo    = (const float*)d_in[14];
  const float* l1s   = (const float*)d_in[15];
  const float* l1b   = (const float*)d_in[16];
  const float* W1    = (const float*)d_in[17];
  const float* b1    = (const float*)d_in[18];
  const float* W2    = (const float*)d_in[19];
  const float* b2    = (const float*)d_in[20];
  const float* l2s   = (const float*)d_in[21];
  const float* l2b   = (const float*)d_in[22];
  const float* gWih  = (const float*)d_in[23];
  const float* gWhh  = (const float*)d_in[24];
  const float* gbih  = (const float*)d_in[25];
  const float* gbhh  = (const float*)d_in[26];
  const float* ih    = (const float*)d_in[27];
  const float* fcnW  = (const float*)d_in[28];
  const float* fcnb  = (const float*)d_in[29];
  (void)in_sizes; (void)n_in; (void)out_size; (void)ws_size;

  char* ws = (char*)d_ws;
  float*          x    = (float*)(ws + OFF_X);
  float*          q    = (float*)(ws + OFF_Q);
  float*          k    = (float*)(ws + OFF_K);
  float*          v    = (float*)(ws + OFF_V);
  float*          o    = (float*)(ws + OFF_O);
  float*          gi   = (float*)(ws + OFF_GI);
  float*          gh   = (float*)(ws + OFF_GH);
  float*          h    = (float*)(ws + OFF_H);
  unsigned short* ysw  = (unsigned short*)(ws + OFF_YBF);
  unsigned short* WihSW= (unsigned short*)(ws + OFF_WIHT);
  unsigned short* WhhSW= (unsigned short*)(ws + OFF_WHHT);
  unsigned short* fwsw = (unsigned short*)(ws + OFF_FCNW);
  unsigned short* hsw  = (unsigned short*)(ws + OFF_HBF);

  // weight prep (bf16, fragment-swizzled) — deterministic, every call
  k_conv_gru<<<(G3_ * HID_ + 255) / 256, 256, 0, stream>>>(gWih, gWhh, WihSW, WhhSW);
  k_conv_fcn<<<(HID_ * F_ + 255) / 256, 256, 0, stream>>>(fcnW, fwsw);

  // AutoDis + positional embedding
  k_autodis<<<(W_ * B_ * F_ + 255) / 256, 256, 0, stream>>>(src, w1w, w1b, w2w, w2b, meta, pos, x);

  // transformer encoder
  for (int l = 0; l < NL_; ++l) {
    k_qkv<<<(B_ * S_ + 255) / 256, 256, 0, stream>>>(
        x, Wq + l * E_ * E_, Wk + l * E_ * E_, Wv + l * E_ * E_,
        bq + l * E_, bk + l * E_, bv + l * E_, q, k, v);
    k_attn<<<B_ * NH_ * (S_ / 16), 256, 16 * S_ * sizeof(float), stream>>>(q, k, v, o);
    k_post<<<(B_ * S_ + 255) / 256, 256, 0, stream>>>(
        x, o, Wo + l * E_ * E_, bo + l * E_,
        l1s + l * E_, l1b + l * E_,
        W1 + l * E_ * FFD_, b1 + l * FFD_,
        W2 + l * FFD_ * E_, b2 + l * E_,
        l2s + l * E_, l2b + l * E_);
  }

  // GRU over first W-1 = 8 steps
  k_pack_y<<<((W_ - 1) * B_ * HID_ + 255) / 256, 256, 0, stream>>>(x, ysw);
  k_init_h<<<(B_ * HID_ + 255) / 256, 256, 0, stream>>>(ih, h, hsw);
  for (int t = 0; t < W_ - 1; ++t) {
    k_gru_mm<<<G3_ / 16, 32, 0, stream>>>(ysw + (size_t)t * B_ * HID_, hsw,
                                          WihSW, WhhSW, gbih, gbhh, gi, gh);
    k_gru_comb<<<(B_ * HID_ + 255) / 256, 256, 0, stream>>>(gi, gh, h, hsw);
  }

  // final head -> d_out [1,B,F] f32
  k_final<<<F_ / 16, 32, 0, stream>>>(hsw, fwsw, fcnb, (float*)d_out);
}
